// ScatterNet1D_16011638079684
// MI455X (gfx1250) — compile-verified, matrix operand order checked by reading, not run on hardware
//
#include <hip/hip_runtime.h>
#include <hip/hip_bf16.h>

typedef _Float16 half_t;
typedef __attribute__((ext_vector_type(16))) _Float16 v16h;
typedef __attribute__((ext_vector_type(8)))  _Float16 v8h;    // 16B-aligned
typedef __attribute__((ext_vector_type(8), aligned(2))) _Float16 v8h_u;  // unaligned load
typedef __attribute__((ext_vector_type(8)))  float    v8f;

#define T_LEN   2048
#define T_MASK  (T_LEN - 1)
#define LEXT    (T_LEN + 64)       // periodic extension: hr_ext[i] = hr[i & 2047], i in [0, T+64)
#define N_SIG   48                 // B*C = 8*6 signals
#define N1F     80                 // first-order filters (J*Q = 10*8)
#define J_OCT   10
#define N_COEF  440                // 80 + 360
#define SCALE_F 256.0f             // filter pre-scale to avoid f16 denormals
#define INV_SCALE_F (1.0f / 256.0f)
#define LOG_EPS_F 1e-6f
#define WPB 8                      // waves per 256-thread block (wave32)

// ---------------------------------------------------------------- filter build
// Filters are stored TIME-REVERSED: hr[tau] = h[(T - tau) mod T], so circulant
// B-tiles become forward-contiguous loads (no per-element reversal in the GEMM).
__device__ inline void store_ext(half_t* base, int rev, float v) {
    half_t h = (half_t)v;
    base[rev] = h;
    if (rev < 64) base[rev + T_LEN] = h;
}

// h[row, tau] = (SCALE/T) * sum_f psi_row(freq(f)) * exp(+2*pi*i*f*tau/T)
// Exact DFT of the sampled frequency-domain Gaussians (matches jnp.fft.ifft of psi).
__global__ void build_filters(half_t* __restrict__ h1re, half_t* __restrict__ h1im,
                              half_t* __restrict__ h2re, half_t* __restrict__ h2im,
                              half_t* __restrict__ gre) {
    int tid = blockIdx.x * blockDim.x + threadIdx.x;
    if (tid >= (N1F + J_OCT + 1) * T_LEN) return;
    int row = tid / T_LEN, tau = tid % T_LEN;

    float xi, sig;
    if (row < N1F)              { xi = 0.4f * exp2f(-(float)row / 8.0f); sig = 0.1f * xi;  } // 0.8*xi/Q
    else if (row < N1F + J_OCT) { xi = 0.4f * exp2f(-(float)(row - N1F)); sig = 0.8f * xi; }
    else                        { xi = 0.0f; sig = 0.8f * 0.4f * exp2f(-10.0f); }            // phi

    const float step = 6.28318530717958647692f * (float)tau / (float)T_LEN;
    float cs, sn;
    sincosf(step, &sn, &cs);
    float accre = 0.0f, accim = 0.0f;
    float cr = 1.0f, ci = 0.0f;
    float inv2s2 = 1.0f / (2.0f * sig * sig);
    for (int f = 0; f < T_LEN; ++f) {
        if ((f & 127) == 0) {  // integer-exact phase resync: 2*pi*((tau*f) mod T)/T
            int m = (tau * f) & T_MASK;
            float a = 6.28318530717958647692f * (float)m / (float)T_LEN;
            float s2, c2; sincosf(a, &s2, &c2);
            cr = c2; ci = s2;
        }
        float fr = ((f < T_LEN / 2) ? (float)f : (float)(f - T_LEN)) / (float)T_LEN;
        float d  = fr - xi;
        float p  = __expf(-d * d * inv2s2);   // v_exp_f32; error << f16 storage quantization
        accre += p * cr;
        accim += p * ci;
        float nr = cr * cs - ci * sn;  // rotate phasor by step
        ci = cr * sn + ci * cs;
        cr = nr;
    }
    accre *= SCALE_F / (float)T_LEN;
    accim *= SCALE_F / (float)T_LEN;

    int rev = (T_LEN - tau) & T_MASK;   // time-reversed storage index
    if (row < N1F) {
        store_ext(h1re + (size_t)row * LEXT, rev, accre);
        store_ext(h1im + (size_t)row * LEXT, rev, accim);
    } else if (row < N1F + J_OCT) {
        int k = row - N1F;
        store_ext(h2re + (size_t)k * LEXT, rev, accre);
        store_ext(h2im + (size_t)k * LEXT, rev, accim);
    } else {
        store_ext(gre, rev, accre);  // phi is symmetric -> real lowpass
    }
}

// ---------------------------------------------------------------- input cast
// x: [8][2048][6] f32  ->  xf: [48][2048] f16, row n = b*6 + c
__global__ void cast_x(const float* __restrict__ x, half_t* __restrict__ xf) {
    int tid = blockIdx.x * blockDim.x + threadIdx.x;
    if (tid >= N_SIG * T_LEN) return;
    int n = tid / T_LEN, t = tid % T_LEN;
    int b = n / 6, c = n % 6;
    xf[tid] = (half_t)x[((size_t)b * T_LEN + t) * 6 + c];
}

__global__ void zero_acc(float* __restrict__ acc) {
    int tid = blockIdx.x * blockDim.x + threadIdx.x;
    if (tid < N_SIG * N_COEF) acc[tid] = 0.0f;
}

// ---------------------------------------------------------------- WMMA tile loads
// A 16x32 f16 tile (ISA layout): lane L: M = L%16, g = L/16;
// v0..3 <- K = g*8 + 0..7 ; v4..7 <- K = 16 + g*8 + 0..7  (two contiguous 16B chunks)
__device__ inline v16h load_a(const half_t* aptr /* row base + g*8 + kk */) {
    v8h lo = *(const v8h*)(aptr);
    v8h hi = *(const v8h*)(aptr + 16);
    v16h a;
#pragma unroll
    for (int i = 0; i < 8; ++i) { a[i] = lo[i]; a[i + 8] = hi[i]; }
    return a;
}

// B 32x16 f16 circulant tile with time-reversed filter:
// B[k,n] = hr_ext[(kk + k - tb - n) & 2047]; lane L: n = L%16, g = L/16.
// Same K pattern as A -> two forward-contiguous 16B chunks (2-byte aligned).
__device__ inline v16h load_b(const half_t* __restrict__ hext, int base /* (kk + g*8 - tb - nn) & 2047 */) {
    v8h_u lo = *(const v8h_u*)(hext + base);
    v8h_u hi = *(const v8h_u*)(hext + base + 16);
    v16h b;
#pragma unroll
    for (int i = 0; i < 8; ++i) { b[i] = lo[i]; b[i + 8] = hi[i]; }
    return b;
}

// ---------------------------------------------------------------- conv + |.| stages
// MODE 0 (order 1): A = xf [48][T], filter j = per-wave, out row = n*80 + j  (U1)
// MODE 1 (order 2): A = U1, rows m = n*div + j1 with A row n*80 + j1, filter fixed (h2[k]), out row = m (U2 slab)
template <int MODE>
__global__ void conv_abs(const half_t* __restrict__ A,
                         const half_t* __restrict__ hre, const half_t* __restrict__ him,
                         half_t* __restrict__ U, int nwaves, int div) {
    int lane = threadIdx.x & 31;
    int wid  = blockIdx.x * WPB + (threadIdx.x >> 5);
    if (wid >= nwaves) return;

    int strip = wid & 31;        // 32 strips of 64 output columns
    int rest  = wid >> 5;
    int mt, jfilt = 0;
    const half_t *hre_f, *him_f;
    if constexpr (MODE == 0) {
        mt = rest % 3; jfilt = rest / 3;
        hre_f = hre + (size_t)jfilt * LEXT;
        him_f = him + (size_t)jfilt * LEXT;
    } else {
        mt = rest; hre_f = hre; him_f = him;
    }

    int nn = lane & 15, g = lane >> 4;
    int m_a = mt * 16 + nn;                      // A-matrix row for this lane
    size_t arow;
    if constexpr (MODE == 0) arow = (size_t)m_a;
    else { int n = m_a / div, j1 = m_a % div; arow = (size_t)(n * N1F + j1); }
    const half_t* aptr = A + arow * T_LEN + g * 8;

    int t0 = strip * 64;
    int c0 = g * 8 - nn - t0;    // base = (c0 + kk - nt*16) & T_MASK
    v8f zero = {0.f, 0.f, 0.f, 0.f, 0.f, 0.f, 0.f, 0.f};
    v8f cre[4], cim[4];
#pragma unroll
    for (int nt = 0; nt < 4; ++nt) { cre[nt] = zero; cim[nt] = zero; }

    for (int kk = 0; kk < T_LEN; kk += 32) {
        __builtin_prefetch(aptr + kk + 64, 0, 3);   // WGP-scope prefetch of streamed A row
        v16h a = load_a(aptr + kk);
#pragma unroll
        for (int nt = 0; nt < 4; ++nt) {
            int base = (c0 + kk - nt * 16) & T_MASK;
            v16h bre = load_b(hre_f, base);
            v16h bim = load_b(him_f, base);
            cre[nt] = __builtin_amdgcn_wmma_f32_16x16x32_f16(false, a, false, bre, (short)0, cre[nt], false, false);
            cim[nt] = __builtin_amdgcn_wmma_f32_16x16x32_f16(false, a, false, bim, (short)0, cim[nt], false, false);
        }
    }

    // D layout: VGPR v -> row mt*16 + v + 8*g, lane%16 -> column
#pragma unroll
    for (int nt = 0; nt < 4; ++nt) {
        int tb = t0 + nt * 16;
#pragma unroll
        for (int v = 0; v < 8; ++v) {
            int mrow = mt * 16 + v + 8 * g;
            float re = cre[nt][v], im = cim[nt][v];
            float u  = sqrtf(re * re + im * im) * INV_SCALE_F;
            size_t orow = (MODE == 0) ? ((size_t)mrow * N1F + jfilt) : (size_t)mrow;
            U[orow * T_LEN + tb + nn] = (half_t)u;
        }
    }
}

// ---------------------------------------------------------------- smoothing + log-mean reduce
// S[r] = A[r] (.) g ; contributes sum_t log(S + eps) into acc[n*440 + qbase + (r % rows_per_n)]
__global__ void conv_smooth_reduce(const half_t* __restrict__ A, const half_t* __restrict__ gext,
                                   float* __restrict__ acc, int nwaves, int rows_per_n, int qbase) {
    int lane = threadIdx.x & 31;
    int wid  = blockIdx.x * WPB + (threadIdx.x >> 5);
    if (wid >= nwaves) return;

    int strip = wid & 31;
    int mt    = wid >> 5;
    int nn = lane & 15, g = lane >> 4;
    int m_a = mt * 16 + nn;
    const half_t* aptr = A + (size_t)m_a * T_LEN + g * 8;

    int t0 = strip * 64;
    int c0 = g * 8 - nn - t0;
    v8f zero = {0.f, 0.f, 0.f, 0.f, 0.f, 0.f, 0.f, 0.f};
    v8f cc[4];
#pragma unroll
    for (int nt = 0; nt < 4; ++nt) cc[nt] = zero;

    for (int kk = 0; kk < T_LEN; kk += 32) {
        __builtin_prefetch(aptr + kk + 64, 0, 3);
        v16h a = load_a(aptr + kk);
#pragma unroll
        for (int nt = 0; nt < 4; ++nt) {
            int base = (c0 + kk - nt * 16) & T_MASK;
            v16h b = load_b(gext, base);
            cc[nt] = __builtin_amdgcn_wmma_f32_16x16x32_f16(false, a, false, b, (short)0, cc[nt], false, false);
        }
    }

    // per-row sum of log over the wave's 64 columns, then atomic into acc.
    // Argument of log is >= 1e-6 (never denormal) -> fast __logf (bare v_log_f32) is exact enough.
#pragma unroll
    for (int v = 0; v < 8; ++v) {
        float l = 0.0f;
#pragma unroll
        for (int nt = 0; nt < 4; ++nt) {
            float s = cc[nt][v] * INV_SCALE_F;
            l += __logf(s + LOG_EPS_F);
        }
        // reduce across the 16 lanes of this half (same output row, 16 columns each)
        for (int off = 1; off < 16; off <<= 1) l += __shfl_xor(l, off, 16);
        if (nn == 0) {
            int r = mt * 16 + v + 8 * g;
            int n = r / rows_per_n;
            int q = qbase + (r % rows_per_n);
            atomicAdd(&acc[n * N_COEF + q], l);
        }
    }
}

// ---------------------------------------------------------------- output
// d_out[b, q, c] = acc[(b*6+c)*440 + q] / T
__global__ void finalize(const float* __restrict__ acc, float* __restrict__ out) {
    int tid = blockIdx.x * blockDim.x + threadIdx.x;
    if (tid >= 8 * N_COEF * 6) return;
    int c = tid % 6;
    int q = (tid / 6) % N_COEF;
    int b = tid / (6 * N_COEF);
    out[tid] = acc[(b * 6 + c) * N_COEF + q] * (1.0f / (float)T_LEN);
}

// ---------------------------------------------------------------- launch
extern "C" void kernel_launch(void* const* d_in, const int* in_sizes, int n_in,
                              void* d_out, int out_size, void* d_ws, size_t ws_size,
                              hipStream_t stream) {
    (void)in_sizes; (void)n_in; (void)out_size; (void)ws_size;
    const float* x = (const float*)d_in[0];
    float* out = (float*)d_out;

    char* ws = (char*)d_ws;
    auto carve = [&](size_t bytes) {
        char* p = ws;
        ws += (bytes + 255) & ~(size_t)255;
        return p;
    };
    half_t* xf   = (half_t*)carve((size_t)N_SIG * T_LEN * 2);
    half_t* h1re = (half_t*)carve((size_t)N1F * LEXT * 2);
    half_t* h1im = (half_t*)carve((size_t)N1F * LEXT * 2);
    half_t* h2re = (half_t*)carve((size_t)J_OCT * LEXT * 2);
    half_t* h2im = (half_t*)carve((size_t)J_OCT * LEXT * 2);
    half_t* gext = (half_t*)carve((size_t)LEXT * 2);
    half_t* U1   = (half_t*)carve((size_t)N_SIG * N1F * T_LEN * 2);    // 15.7 MB
    half_t* U2   = (half_t*)carve((size_t)N_SIG * 72 * T_LEN * 2);     // 14.2 MB (max slab, k=9)
    float*  acc  = (float*)carve((size_t)N_SIG * N_COEF * 4);

    cast_x<<<(N_SIG * T_LEN + 255) / 256, 256, 0, stream>>>(x, xf);
    build_filters<<<((N1F + J_OCT + 1) * T_LEN + 255) / 256, 256, 0, stream>>>(h1re, h1im, h2re, h2im, gext);
    zero_acc<<<(N_SIG * N_COEF + 255) / 256, 256, 0, stream>>>(acc);

    // order 1: 80 filters x 3 M-tiles (48 rows) x 32 strips
    {
        int nw = N1F * 3 * 32;
        conv_abs<0><<<nw / WPB, 256, 0, stream>>>(xf, h1re, h1im, U1, nw, 1);
    }
    // S1 smoothing + log-mean: 3840 rows -> 240 M-tiles x 32 strips
    {
        int nw = 240 * 32;
        conv_smooth_reduce<<<nw / WPB, 256, 0, stream>>>(U1, gext, acc, nw, N1F, 0);
    }
    // order 2: per octave k, pairs j1 in [0, 8k)
    for (int k = 1; k < J_OCT; ++k) {
        int rows = N_SIG * 8 * k;       // multiple of 16
        int nw = (rows / 16) * 32;
        conv_abs<1><<<nw / WPB, 256, 0, stream>>>(U1, h2re + (size_t)k * LEXT, h2im + (size_t)k * LEXT,
                                                  U2, nw, 8 * k);
        conv_smooth_reduce<<<nw / WPB, 256, 0, stream>>>(U2, gext, acc, nw, 8 * k,
                                                         N1F + 4 * k * (k - 1));
    }
    finalize<<<(8 * N_COEF * 6 + 255) / 256, 256, 0, stream>>>(acc, out);
}
